// GRULayer_31671088841298
// MI455X (gfx1250) — matrix operand order repaired and checked by reading, not compiled, not run on hardware
//
#include <hip/hip_runtime.h>

// GRU layer: B=64, T=512, D=256, H=512
#define Bsz 64
#define Tsz 512
#define Dsz 256
#define Hsz 512
#define NCOLS 1536          // r|u|c concatenated output columns
#define NCP   1540          // padded G row (1540 % 64 == 4 -> conflict-free half-wave stores)
#define HBP   520           // padded hb row in bf16 elements (520*2B = 1040B, 16B-aligned)
#define NT    96            // NCOLS / 16 N-tiles
#define TPB   1024          // 32 waves (wave32), 8 waves/SIMD
#define TPW   3             // N-tiles per wave: 32 waves * 3 = 96

typedef __attribute__((ext_vector_type(16))) __bf16 v16bf;
typedef __attribute__((ext_vector_type(8)))  float  v8f;

union FragAB { uint4 u[2]; v16bf v; };

__device__ __forceinline__ float sigmoidf_(float x) { return 1.0f / (1.0f + expf(-x)); }

// ---------------- workspace layout (bytes) ----------------
#define OFF_XB 0ull                              // x as bf16: 32768*256*2        = 16 MB
#define OFF_WX (OFF_XB + 16777216ull)            // packed W_x frags: 8*96*1024   = 768 KB
#define OFF_WH (OFF_WX + 786432ull)              // packed W_h frags: 16*96*1024  = 1.5 MB
#define OFF_XR (OFF_WH + 1572864ull)             // LN(x@Wxr): 32768*512*4        = 64 MB
#define OFF_XU (OFF_XR + 67108864ull)
#define OFF_XC (OFF_XU + 67108864ull)            // total ~210 MB

// ---------------- f32 -> bf16 convert ----------------
__global__ void k_f32_to_bf16(const float* __restrict__ src, __bf16* __restrict__ dst, int n) {
    int i = blockIdx.x * blockDim.x + threadIdx.x;
    if (i < n) dst[i] = (__bf16)src[i];
}

// ---------------- repack W (K x 1536, r|u|c) into WMMA B-fragment order ----------------
// Packed element e: tile = e>>9 (kt = tile/96, nt = tile%96), p = e&511, lane = p>>4, i = p&15.
// B 32x16 bf16 fragment (contiguous k order used here, same as A): lanes 0-15 -> K = kb+0..15, lanes 16-31 -> K = kb+16..31,
// column n = lane&15, element i = K offset within the lane's 16-row span.
__global__ void k_pack_w(const float* __restrict__ wr, const float* __restrict__ wu,
                         const float* __restrict__ wc, __bf16* __restrict__ dst, int total) {
    int e = blockIdx.x * blockDim.x + threadIdx.x;
    if (e >= total) return;
    int p    = e & 511;
    int tile = e >> 9;
    int kt = tile / NT, nt = tile % NT;
    int lane = p >> 4, i = p & 15;
    int k = kt * 32 + (lane >> 4) * 16 + i;
    int n = nt * 16 + (lane & 15);
    float v;
    if (n < Hsz)            v = wr[k * Hsz + n];
    else if (n < 2 * Hsz)   v = wu[k * Hsz + (n - Hsz)];
    else                    v = wc[k * Hsz + (n - 2 * Hsz)];
    dst[e] = (__bf16)v;
}

// A 16x32 bf16 fragment: lanes 0-15 row=lane, K=kb..kb+15 (contiguous);
//                        lanes 16-31 row=lane-16, K=kb+16..kb+31.
// C 16x16 f32 fragment:  element r -> (row = r + (lane>>4)*8, col = lane&15).

// ---------------- Phase 1: x projection + LN (fused) ----------------
__global__ __launch_bounds__(TPB) void k_xproj(
    const __bf16* __restrict__ xb, const __bf16* __restrict__ wxp,
    const float* __restrict__ ln_ru_s, const float* __restrict__ ln_ru_b,
    const float* __restrict__ ln_c_s,  const float* __restrict__ ln_c_b,
    float* __restrict__ xr, float* __restrict__ xu, float* __restrict__ xc)
{
    __shared__ __align__(16) float G[16][NCP];           // ~96 KB
    __shared__ float s_mru[16], s_iru[16], s_mc[16], s_ic[16];
    const int tid = threadIdx.x, wave = tid >> 5, lane = tid & 31;
    const int half = lane >> 4, l15 = lane & 15;
    const size_t R0 = (size_t)blockIdx.x * 16;           // 16 rows of (B*T) per WG
    const int KT = Dsz / 32;                             // 8

    v8f acc[TPW] = {};
    const __bf16* arow  = xb + (R0 + (size_t)l15) * Dsz;
    const __bf16* bbase = wxp + ((size_t)(wave * TPW) * 32 + lane) * 16;

    FragAB A[2], Bb[2][TPW];
    A[0].u[0] = *(const uint4*)(arow + half * 16);
    A[0].u[1] = *(const uint4*)(arow + half * 16 + 8);
    #pragma unroll
    for (int q = 0; q < TPW; ++q) {
        Bb[0][q].u[0] = *(const uint4*)(bbase + q * 512);
        Bb[0][q].u[1] = *(const uint4*)(bbase + q * 512 + 8);
    }
    #pragma unroll
    for (int kt = 0; kt < KT; ++kt) {
        const int cur = kt & 1, nxt = cur ^ 1;
        if (kt + 1 < KT) {
            const __bf16* an = arow + (kt + 1) * 32;
            A[nxt].u[0] = *(const uint4*)(an + half * 16);
            A[nxt].u[1] = *(const uint4*)(an + half * 16 + 8);
            const __bf16* bn = bbase + (size_t)(kt + 1) * NT * 512;
            #pragma unroll
            for (int q = 0; q < TPW; ++q) {
                Bb[nxt][q].u[0] = *(const uint4*)(bn + q * 512);
                Bb[nxt][q].u[1] = *(const uint4*)(bn + q * 512 + 8);
            }
        }
        if (kt + 2 < KT)
            __builtin_prefetch(bbase + (size_t)(kt + 2) * NT * 512, 0, 1);
        #pragma unroll
        for (int q = 0; q < TPW; ++q)
            acc[q] = __builtin_amdgcn_wmma_f32_16x16x32_bf16(
                false, A[cur].v, false, Bb[cur][q].v, (short)0, acc[q], false, false);
    }
    #pragma unroll
    for (int q = 0; q < TPW; ++q) {
        const int n0 = (wave * TPW + q) * 16 + l15;
        #pragma unroll
        for (int r = 0; r < 8; ++r) G[r + half * 8][n0] = acc[q][r];
    }
    __syncthreads();

    if (wave < 16) {   // LN stats: one wave per row
        const int m = wave;
        float s = 0.f, q = 0.f;
        for (int i = 0; i < 32; ++i) { float v = G[m][lane + i * 32]; s += v; q += v * v; }
        for (int o = 16; o >= 1; o >>= 1) { s += __shfl_xor(s, o, 32); q += __shfl_xor(q, o, 32); }
        float mean = s * (1.0f / 1024.0f), var = q * (1.0f / 1024.0f) - mean * mean;
        if (lane == 0) { s_mru[m] = mean; s_iru[m] = rsqrtf(var + 1e-5f); }
        s = 0.f; q = 0.f;
        for (int i = 0; i < 16; ++i) { float v = G[m][1024 + lane + i * 32]; s += v; q += v * v; }
        for (int o = 16; o >= 1; o >>= 1) { s += __shfl_xor(s, o, 32); q += __shfl_xor(q, o, 32); }
        mean = s * (1.0f / 512.0f); var = q * (1.0f / 512.0f) - mean * mean;
        if (lane == 0) { s_mc[m] = mean; s_ic[m] = rsqrtf(var + 1e-5f); }
    }
    __syncthreads();

    const int j = tid & 511, mh = tid >> 9;              // 1024 threads: 2 half-row groups
    const float sr = ln_ru_s[j],       br = ln_ru_b[j];
    const float su = ln_ru_s[Hsz + j], bu = ln_ru_b[Hsz + j];
    const float sc = ln_c_s[j],        bc = ln_c_b[j];
    #pragma unroll
    for (int i = 0; i < 8; ++i) {
        const int m = mh * 8 + i;
        const size_t o = (R0 + m) * Hsz + j;
        xr[o] = (G[m][j]           - s_mru[m]) * s_iru[m] * sr + br;
        xu[o] = (G[m][Hsz + j]     - s_mru[m]) * s_iru[m] * su + bu;
        xc[o] = (G[m][2 * Hsz + j] - s_mc[m])  * s_ic[m]  * sc + bc;
    }
}

// ---------------- Phase 2: persistent recurrence, 16 batch rows per WG ----------------
__global__ __launch_bounds__(TPB) void k_recur(
    const __bf16* __restrict__ whp,
    const float* __restrict__ xr, const float* __restrict__ xu, const float* __restrict__ xc,
    const float* __restrict__ h0,
    const float* __restrict__ ln_ru_s, const float* __restrict__ ln_ru_b,
    const float* __restrict__ ln_c_s,  const float* __restrict__ ln_c_b,
    float* __restrict__ out)
{
    __shared__ __align__(16) float  G[16][NCP];          // ~96 KB
    __shared__ __align__(16) float  hf[16][Hsz];         // 32 KB fp32 master state
    __shared__ __align__(16) __bf16 hb[16][HBP];         // ~16 KB WMMA A source (padded rows)
    __shared__ float s_mru[16], s_iru[16], s_mc[16], s_ic[16];
    const int tid = threadIdx.x, wave = tid >> 5, lane = tid & 31;
    const int half = lane >> 4, l15 = lane & 15;
    const int b0 = blockIdx.x * 16;
    const int KT = Hsz / 32;                             // 16

    for (int idx = tid; idx < 16 * Hsz; idx += TPB) {
        int m = idx >> 9, jj = idx & (Hsz - 1);
        float v = h0[jj];
        hf[m][jj] = v; hb[m][jj] = (__bf16)v;
    }
    __syncthreads();

    const int j = tid & 511, mh = tid >> 9;
    const float sr = ln_ru_s[j],       br = ln_ru_b[j];
    const float su = ln_ru_s[Hsz + j], bu = ln_ru_b[Hsz + j];
    const float sc = ln_c_s[j],        bc = ln_c_b[j];
    const __bf16* bbase = whp + ((size_t)(wave * TPW) * 32 + lane) * 16;

    for (int t = 0; t < Tsz; ++t) {
        v8f acc[TPW] = {};
        FragAB A[2], Bb[2][TPW];
        A[0].u[0] = *(const uint4*)&hb[l15][half * 16];
        A[0].u[1] = *(const uint4*)&hb[l15][half * 16 + 8];
        #pragma unroll
        for (int q = 0; q < TPW; ++q) {
            Bb[0][q].u[0] = *(const uint4*)(bbase + q * 512);
            Bb[0][q].u[1] = *(const uint4*)(bbase + q * 512 + 8);
        }
        #pragma unroll
        for (int kt = 0; kt < KT; ++kt) {
            const int cur = kt & 1, nxt = cur ^ 1;
            if (kt + 1 < KT) {
                const int kb = (kt + 1) * 32;
                A[nxt].u[0] = *(const uint4*)&hb[l15][kb + half * 16];
                A[nxt].u[1] = *(const uint4*)&hb[l15][kb + half * 16 + 8];
                const __bf16* bn = bbase + (size_t)(kt + 1) * NT * 512;
                #pragma unroll
                for (int q = 0; q < TPW; ++q) {
                    Bb[nxt][q].u[0] = *(const uint4*)(bn + q * 512);
                    Bb[nxt][q].u[1] = *(const uint4*)(bn + q * 512 + 8);
                }
            }
            if (kt + 2 < KT)
                __builtin_prefetch(bbase + (size_t)(kt + 2) * NT * 512, 0, 1);
            #pragma unroll
            for (int q = 0; q < TPW; ++q)
                acc[q] = __builtin_amdgcn_wmma_f32_16x16x32_bf16(
                    false, A[cur].v, false, Bb[cur][q].v, (short)0, acc[q], false, false);
        }
        #pragma unroll
        for (int q = 0; q < TPW; ++q) {
            const int n0 = (wave * TPW + q) * 16 + l15;
            #pragma unroll
            for (int r = 0; r < 8; ++r) G[r + half * 8][n0] = acc[q][r];
        }
        __syncthreads();

        if (wave < 16) {   // LN stats: one wave per row
            const int m = wave;
            float s = 0.f, q = 0.f;
            for (int i = 0; i < 32; ++i) { float v = G[m][lane + i * 32]; s += v; q += v * v; }
            for (int o = 16; o >= 1; o >>= 1) { s += __shfl_xor(s, o, 32); q += __shfl_xor(q, o, 32); }
            float mean = s * (1.0f / 1024.0f), var = q * (1.0f / 1024.0f) - mean * mean;
            if (lane == 0) { s_mru[m] = mean; s_iru[m] = rsqrtf(var + 1e-5f); }
            s = 0.f; q = 0.f;
            for (int i = 0; i < 16; ++i) { float v = G[m][1024 + lane + i * 32]; s += v; q += v * v; }
            for (int o = 16; o >= 1; o >>= 1) { s += __shfl_xor(s, o, 32); q += __shfl_xor(q, o, 32); }
            mean = s * (1.0f / 512.0f); var = q * (1.0f / 512.0f) - mean * mean;
            if (lane == 0) { s_mc[m] = mean; s_ic[m] = rsqrtf(var + 1e-5f); }
        }
        __syncthreads();

        // gates + state update + output (each thread owns (8 rows) x (1 column))
        #pragma unroll
        for (int i = 0; i < 8; ++i) {
            const int m = mh * 8 + i;
            const size_t base = (((size_t)(b0 + m)) * Tsz + t) * Hsz + j;
            float gr = (G[m][j]           - s_mru[m]) * s_iru[m] * sr + br;
            float gu = (G[m][Hsz + j]     - s_mru[m]) * s_iru[m] * su + bu;
            float gc = (G[m][2 * Hsz + j] - s_mc[m])  * s_ic[m]  * sc + bc;
            float r = sigmoidf_(xr[base] + gr);
            float u = sigmoidf_(xu[base] + gu);
            float c = tanhf(xc[base] + r * gc);
            float h = hf[m][j];
            float hn = (1.0f - u) * h + u * c;
            out[base] = hn;
            hf[m][j] = hn;
            hb[m][j] = (__bf16)hn;
        }
        __syncthreads();
    }
}

extern "C" void kernel_launch(void* const* d_in, const int* in_sizes, int n_in,
                              void* d_out, int out_size, void* d_ws, size_t ws_size,
                              hipStream_t stream) {
    const float* x        = (const float*)d_in[0];
    const float* W_xr     = (const float*)d_in[1];
    const float* W_xu     = (const float*)d_in[2];
    const float* W_xc     = (const float*)d_in[3];
    const float* W_hr     = (const float*)d_in[4];
    const float* W_hu     = (const float*)d_in[5];
    const float* W_hc     = (const float*)d_in[6];
    const float* h0       = (const float*)d_in[7];
    const float* ln_xru_s = (const float*)d_in[8];
    const float* ln_xru_b = (const float*)d_in[9];
    const float* ln_hru_s = (const float*)d_in[10];
    const float* ln_hru_b = (const float*)d_in[11];
    const float* ln_xc_s  = (const float*)d_in[12];
    const float* ln_xc_b  = (const float*)d_in[13];
    const float* ln_hc_s  = (const float*)d_in[14];
    const float* ln_hc_b  = (const float*)d_in[15];
    float* out = (float*)d_out;

    char* ws = (char*)d_ws;
    __bf16* xb  = (__bf16*)(ws + OFF_XB);
    __bf16* wxp = (__bf16*)(ws + OFF_WX);
    __bf16* whp = (__bf16*)(ws + OFF_WH);
    float*  xr  = (float*)(ws + OFF_XR);
    float*  xu  = (float*)(ws + OFF_XU);
    float*  xc  = (float*)(ws + OFF_XC);

    // 1. convert x to bf16
    const int nx = Bsz * Tsz * Dsz;
    k_f32_to_bf16<<<nx / 256, 256, 0, stream>>>(x, xb, nx);

    // 2. repack weights into WMMA fragment order (bf16)
    const int nwx = (Dsz / 32) * NT * 512;
    k_pack_w<<<nwx / 256, 256, 0, stream>>>(W_xr, W_xu, W_xc, wxp, nwx);
    const int nwh = (Hsz / 32) * NT * 512;
    k_pack_w<<<nwh / 256, 256, 0, stream>>>(W_hr, W_hu, W_hc, whp, nwh);

    // 3. fused x-projection GEMM + LN  (2048 WGs x 32 waves, 16 rows each)
    k_xproj<<<(Bsz * Tsz) / 16, TPB, 0, stream>>>(
        xb, wxp, ln_xru_s, ln_xru_b, ln_xc_s, ln_xc_b, xr, xu, xc);

    // 4. persistent recurrence (4 WGs x 16 batch rows, 512 steps)
    k_recur<<<Bsz / 16, TPB, 0, stream>>>(
        whp, xr, xu, xc, h0, ln_hru_s, ln_hru_b, ln_hc_s, ln_hc_b, out);
}
